// GCNNet768_8143257993633
// MI455X (gfx1250) — compile-verified, matrix operand order checked by reading, not run on hardware
//
#include <hip/hip_runtime.h>
#include <hip/hip_bf16.h>

typedef __attribute__((ext_vector_type(16))) __bf16 v16bf;
typedef __attribute__((ext_vector_type(8)))  float  v8f;

#define NN 10000
#define NE 160000
#define BN_EPS 1e-5f

static __device__ __forceinline__ unsigned short f2bf(float f) {
    unsigned int u = __float_as_uint(f);
    unsigned int r = (u + 0x7FFFu + ((u >> 16) & 1u)) >> 16;   // round-to-nearest-even
    return (unsigned short)r;
}

// ---------------------------------------------------------------- graph prep
__global__ void k_deg(const int* ei, const float* ew, float* deg, int E) {
    int e = blockIdx.x * blockDim.x + threadIdx.x;
    if (e < E) atomicAdd(&deg[ei[e]], ew[e]);
}

__global__ void k_norm(const int* ei, const float* ew, const float* deg, float* nrm, int E) {
    int e = blockIdx.x * blockDim.x + threadIdx.x;
    if (e >= E) return;
    int s = ei[e], d = ei[E + e];
    float ds = deg[s], dd = deg[d];
    float is = ds > 0.f ? rsqrtf(ds) : 0.f;
    float id = dd > 0.f ? rsqrtf(dd) : 0.f;
    nrm[e] = -is * ew[e] * id;
}

// ---------------------------------------------------------------- conversions
__global__ void k_tobf16(const float* x, unsigned short* y, int n) {
    int i = blockIdx.x * blockDim.x + threadIdx.x;
    if (i < n) y[i] = f2bf(x[i]);
}

// W: K x Ncol (row-major f32)  ->  Wt: Ncol x K (row-major bf16)
__global__ void k_convT(const float* W, unsigned short* Wt, int K, int Ncol) {
    int i = blockIdx.x * blockDim.x + threadIdx.x;
    if (i >= K * Ncol) return;
    int k = i / Ncol, n = i % Ncol;
    Wt[(size_t)n * K + k] = f2bf(W[i]);
}

// ---------------------------------------------------------------- WMMA GEMM
// C[M x Ncol] (+)= A[M x K](bf16) * Bt[Ncol x K](bf16, pre-transposed)
// One wave computes a 16 x 64 strip (4 accumulators). Fragments are
// double-buffered one K-step ahead so loads overlap the WMMAs.
union Frag { v16bf v; uint4 q[2]; };

__global__ void k_gemm(const unsigned short* __restrict__ A,
                       const unsigned short* __restrict__ Bt,
                       float* __restrict__ C,
                       int M, int Ncol, int K, int acc) {
    const int wave   = blockIdx.x * (blockDim.x >> 5) + (threadIdx.x >> 5);
    const int lane   = threadIdx.x & 31;
    const int tilesN = Ncol >> 6;                 // 64-wide strips
    const int tilesM = M >> 4;
    if (wave >= tilesM * tilesN) return;          // wave-uniform: EXEC stays all-ones
    const int tm = wave / tilesN, tn = wave % tilesN;
    const int m  = tm * 16 + (lane & 15);         // A row for this lane
    const int nb = tn * 64 + (lane & 15);         // B col (first sub-tile)
    const int kb = (lane >> 4) << 3;              // K sub-offset: 0 or 8

    const unsigned short* arow = A + (size_t)m * K + kb;
    const unsigned short* brow = Bt + (size_t)nb * K + kb;
    const size_t bstep = (size_t)16 * K;          // 16 columns ahead

    v8f c0 = {}, c1 = {}, c2 = {}, c3 = {};
    const int rowb = tm * 16 + ((lane >> 4) << 3);
    if (acc) {
#pragma unroll
        for (int r = 0; r < 8; ++r) {
            const float* cr = C + (size_t)(rowb + r) * Ncol + tn * 64 + (lane & 15);
            c0[r] = cr[0]; c1[r] = cr[16]; c2[r] = cr[32]; c3[r] = cr[48];
        }
    }

    Frag a, b0, b1, b2, b3;
    Frag an, bn0, bn1, bn2, bn3;

    // prologue: load K-step 0
    a.q[0]  = *(const uint4*)(arow);
    a.q[1]  = *(const uint4*)(arow + 16);
    b0.q[0] = *(const uint4*)(brow);
    b0.q[1] = *(const uint4*)(brow + 16);
    b1.q[0] = *(const uint4*)(brow + bstep);
    b1.q[1] = *(const uint4*)(brow + bstep + 16);
    b2.q[0] = *(const uint4*)(brow + 2 * bstep);
    b2.q[1] = *(const uint4*)(brow + 2 * bstep + 16);
    b3.q[0] = *(const uint4*)(brow + 3 * bstep);
    b3.q[1] = *(const uint4*)(brow + 3 * bstep + 16);

    for (int kk = 0; kk < K; kk += 32) {
        // issue next K-step's loads before consuming the current fragments
        const int kn = (kk + 32 < K) ? (kk + 32) : kk;   // last iter: harmless reload
        an.q[0]  = *(const uint4*)(arow + kn);
        an.q[1]  = *(const uint4*)(arow + kn + 16);
        bn0.q[0] = *(const uint4*)(brow + kn);
        bn0.q[1] = *(const uint4*)(brow + kn + 16);
        bn1.q[0] = *(const uint4*)(brow + bstep + kn);
        bn1.q[1] = *(const uint4*)(brow + bstep + kn + 16);
        bn2.q[0] = *(const uint4*)(brow + 2 * bstep + kn);
        bn2.q[1] = *(const uint4*)(brow + 2 * bstep + kn + 16);
        bn3.q[0] = *(const uint4*)(brow + 3 * bstep + kn);
        bn3.q[1] = *(const uint4*)(brow + 3 * bstep + kn + 16);

        c0 = __builtin_amdgcn_wmma_f32_16x16x32_bf16(false, a.v, false, b0.v, (short)0, c0, false, false);
        c1 = __builtin_amdgcn_wmma_f32_16x16x32_bf16(false, a.v, false, b1.v, (short)0, c1, false, false);
        c2 = __builtin_amdgcn_wmma_f32_16x16x32_bf16(false, a.v, false, b2.v, (short)0, c2, false, false);
        c3 = __builtin_amdgcn_wmma_f32_16x16x32_bf16(false, a.v, false, b3.v, (short)0, c3, false, false);

        a = an; b0 = bn0; b1 = bn1; b2 = bn2; b3 = bn3;
    }

#pragma unroll
    for (int r = 0; r < 8; ++r) {
        float* cr = C + (size_t)(rowb + r) * Ncol + tn * 64 + (lane & 15);
        cr[0] = c0[r]; cr[16] = c1[r]; cr[32] = c2[r]; cr[48] = c3[r];
    }
}

// ---------------------------------------------------------------- batch norm
// Pass 1: coalesced row-block partial sums (one column per thread).
__global__ void k_bn_partial(const float* __restrict__ X, float* __restrict__ sum,
                             float* __restrict__ ssq, int M, int F, int rowsPerBlock) {
    const int c = blockIdx.y * 256 + threadIdx.x;
    if (c >= F) return;
    const int r0 = blockIdx.x * rowsPerBlock;
    const int r1 = min(M, r0 + rowsPerBlock);
    float s = 0.f, q = 0.f;
    for (int r = r0; r < r1; ++r) {
        float v = X[(size_t)r * F + c];
        s += v; q += v * v;
    }
    atomicAdd(&sum[c], s);
    atomicAdd(&ssq[c], q);
}

__global__ void k_bn_final(const float* sum, const float* ssq, float* mu, float* rs, int M, int F) {
    int c = blockIdx.x * blockDim.x + threadIdx.x;
    if (c >= F) return;
    float m = sum[c] / (float)M;
    float v = ssq[c] / (float)M - m * m;
    mu[c] = m; rs[c] = rsqrtf(v + BN_EPS);
}

__global__ void k_bn_apply(const float* __restrict__ X, float* __restrict__ Y,
                           unsigned short* __restrict__ Yb,
                           const float* mu, const float* rs,
                           const float* g, const float* be,
                           int M, int F, int relu) {
    size_t i = (size_t)blockIdx.x * blockDim.x + threadIdx.x;
    if (i >= (size_t)M * F) return;
    int c = (int)(i % F);
    float v = (X[i] - mu[c]) * rs[c] * g[c] + be[c];
    if (relu) v = fmaxf(v, 0.f);
    Y[i] = v;
    if (Yb) Yb[i] = f2bf(v);
}

// ---------------------------------------------------------------- scatter prop
// O[dst] += nrm[e] * H[src]   (block per edge, threads across features)
__global__ void k_prop(const int* __restrict__ ei, const float* __restrict__ nrm,
                       const float* __restrict__ H, float* __restrict__ O, int F) {
    const int e = blockIdx.x;
    const int s = ei[e], d = ei[NE + e];
    const float w = nrm[e];
    const float* hs = H + (size_t)s * F;
    float* od = O + (size_t)d * F;
    for (int f = threadIdx.x; f < F; f += blockDim.x)
        atomicAdd(&od[f], w * hs[f]);
}

// t2 = 2*q - t0  (+ bf16 copy)
__global__ void k_combine(const float* __restrict__ Q, const float* __restrict__ T0,
                          float* __restrict__ Y, unsigned short* __restrict__ Yb, size_t n) {
    size_t i = (size_t)blockIdx.x * blockDim.x + threadIdx.x;
    if (i >= n) return;
    float v = 2.f * Q[i] - T0[i];
    Y[i] = v; Yb[i] = f2bf(v);
}

// ---------------------------------------------------------------- FC + log_softmax
__global__ void k_fc(const float* __restrict__ H, const float* __restrict__ Wf,
                     const float* __restrict__ bf, float* __restrict__ out, int M) {
    int n = blockIdx.x * blockDim.x + threadIdx.x;
    if (n >= M) return;
    const float* h = H + (size_t)n * 128;
    float z[6], mx = -1e30f;
#pragma unroll
    for (int j = 0; j < 6; ++j) {
        float s = bf[j];
        for (int k = 0; k < 128; ++k) s += h[k] * Wf[k * 6 + j];
        z[j] = s; mx = fmaxf(mx, s);
    }
    float se = 0.f;
#pragma unroll
    for (int j = 0; j < 6; ++j) se += expf(z[j] - mx);
    float l = logf(se);
#pragma unroll
    for (int j = 0; j < 6; ++j) out[(size_t)n * 6 + j] = z[j] - mx - l;
}

// ---------------------------------------------------------------- launch
static inline int gemm_grid(int M, int Ncol) {
    int waves = (M >> 4) * (Ncol >> 6);
    return (waves + 7) / 8;                 // 8 waves (256 threads) per block
}

#define BN_ROWS 125                         // 10000/125 = 80 row-blocks

static inline void run_bn(const float* X, float* Y, unsigned short* Yb,
                          float* bsum, float* bssq, float* mu, float* rs,
                          const float* g, const float* be, int F, int relu,
                          hipStream_t stream) {
    hipMemsetAsync(bsum, 0, (size_t)F * 4, stream);
    hipMemsetAsync(bssq, 0, (size_t)F * 4, stream);
    dim3 gs(NN / BN_ROWS, (F + 255) / 256);
    k_bn_partial<<<gs, 256, 0, stream>>>(X, bsum, bssq, NN, F, BN_ROWS);
    k_bn_final<<<(F + 255) / 256, 256, 0, stream>>>(bsum, bssq, mu, rs, NN, F);
    k_bn_apply<<<((size_t)NN * F + 255) / 256, 256, 0, stream>>>(X, Y, Yb, mu, rs, g, be, NN, F, relu);
}

extern "C" void kernel_launch(void* const* d_in, const int* in_sizes, int n_in,
                              void* d_out, int out_size, void* d_ws, size_t ws_size,
                              hipStream_t stream) {
    const float* x   = (const float*)d_in[0];
    const int*   ei  = (const int*)d_in[1];
    const float* ew  = (const float*)d_in[2];
    const float* W1  = (const float*)d_in[3];
    const float* g1  = (const float*)d_in[5];
    const float* be1 = (const float*)d_in[6];
    const float* W20 = (const float*)d_in[7];
    const float* W21 = (const float*)d_in[8];
    const float* g2  = (const float*)d_in[10];
    const float* be2 = (const float*)d_in[11];
    const float* W30 = (const float*)d_in[12];
    const float* W31 = (const float*)d_in[13];
    const float* W32 = (const float*)d_in[14];
    const float* g3  = (const float*)d_in[16];
    const float* be3 = (const float*)d_in[17];
    const float* fcW = (const float*)d_in[18];
    const float* fcb = (const float*)d_in[19];
    float* out = (float*)d_out;

    char* ws = (char*)d_ws;
    size_t off = 0;
    auto alloc = [&](size_t bytes) -> void* {
        void* p = ws + off;
        off = (off + bytes + 255) & ~(size_t)255;
        return p;
    };

    float*          deg  = (float*)alloc((size_t)NN * 4);
    float*          nrm  = (float*)alloc((size_t)NE * 4);
    float*          mu   = (float*)alloc(512 * 4);
    float*          rs   = (float*)alloc(512 * 4);
    float*          bsum = (float*)alloc(512 * 4);
    float*          bssq = (float*)alloc(512 * 4);
    unsigned short* xb   = (unsigned short*)alloc((size_t)NN * 768 * 2);
    unsigned short* W1t  = (unsigned short*)alloc((size_t)768 * 512 * 2);
    unsigned short* W20t = (unsigned short*)alloc((size_t)512 * 256 * 2);
    unsigned short* W21t = (unsigned short*)alloc((size_t)512 * 256 * 2);
    unsigned short* W30t = (unsigned short*)alloc((size_t)256 * 128 * 2);
    unsigned short* W31t = (unsigned short*)alloc((size_t)256 * 128 * 2);
    unsigned short* W32t = (unsigned short*)alloc((size_t)256 * 128 * 2);
    float*          h1   = (float*)alloc((size_t)NN * 512 * 4);
    unsigned short* h1b  = (unsigned short*)alloc((size_t)NN * 512 * 2);
    float*          p1   = (float*)alloc((size_t)NN * 512 * 4);
    unsigned short* p1b  = (unsigned short*)alloc((size_t)NN * 512 * 2);
    float*          t2   = (float*)alloc((size_t)NN * 256 * 4);

    // Aliases into dead regions (lifetimes are strictly ordered below):
    float*          h2  = h1;                                                   // h1 f32 dead after prop
    unsigned short* h2b = h1b;                                                  // h1b dead after L2 GEMMs
    unsigned short* p2b = (unsigned short*)((char*)h1b + (size_t)NN * 256 * 2);
    float*          p2  = p1;                                                   // p1 f32 dead after p1b conv
    float*          q   = (float*)((char*)p1 + (size_t)NN * 256 * 4);
    unsigned short* t2b = p1b;                                                  // p1b dead after L2 GEMM #2
    float*          h3  = (float*)((char*)p1b + (size_t)NN * 256 * 2);

    const int T = 256;

    // ---- graph normalization
    hipMemsetAsync(deg, 0, (size_t)NN * 4, stream);
    k_deg <<<(NE + T - 1) / T, T, 0, stream>>>(ei, ew, deg, NE);
    k_norm<<<(NE + T - 1) / T, T, 0, stream>>>(ei, ew, deg, nrm, NE);

    // ---- bf16 operand prep
    k_tobf16<<<((NN * 768) + T - 1) / T, T, 0, stream>>>(x, xb, NN * 768);
    k_convT<<<((768 * 512) + T - 1) / T, T, 0, stream>>>(W1,  W1t,  768, 512);
    k_convT<<<((512 * 256) + T - 1) / T, T, 0, stream>>>(W20, W20t, 512, 256);
    k_convT<<<((512 * 256) + T - 1) / T, T, 0, stream>>>(W21, W21t, 512, 256);
    k_convT<<<((256 * 128) + T - 1) / T, T, 0, stream>>>(W30, W30t, 256, 128);
    k_convT<<<((256 * 128) + T - 1) / T, T, 0, stream>>>(W31, W31t, 256, 128);
    k_convT<<<((256 * 128) + T - 1) / T, T, 0, stream>>>(W32, W32t, 256, 128);

    // ---- layer 1: h1 = x @ W1 ; BN ; ReLU      (bias cancels in BN)
    k_gemm<<<gemm_grid(NN, 512), T, 0, stream>>>(xb, W1t, h1, NN, 512, 768, 0);
    run_bn(h1, h1, h1b, bsum, bssq, mu, rs, g1, be1, 512, 1, stream);

    // ---- layer 2: h2 = h1@W2_0 + prop(h1)@W2_1 ; BN ; ReLU
    hipMemsetAsync(p1, 0, (size_t)NN * 512 * 4, stream);
    k_prop<<<NE, T, 0, stream>>>(ei, nrm, h1, p1, 512);
    k_tobf16<<<((NN * 512) + T - 1) / T, T, 0, stream>>>(p1, p1b, NN * 512);
    k_gemm<<<gemm_grid(NN, 256), T, 0, stream>>>(h1b, W20t, h2, NN, 256, 512, 0);
    k_gemm<<<gemm_grid(NN, 256), T, 0, stream>>>(p1b, W21t, h2, NN, 256, 512, 1);
    run_bn(h2, h2, h2b, bsum, bssq, mu, rs, g2, be2, 256, 1, stream);

    // ---- layer 3: Tx1 = prop(h2) ; Tx2 = 2*prop(Tx1) - h2
    hipMemsetAsync(p2, 0, (size_t)NN * 256 * 4, stream);
    k_prop<<<NE, T, 0, stream>>>(ei, nrm, h2, p2, 256);
    k_tobf16<<<((NN * 256) + T - 1) / T, T, 0, stream>>>(p2, p2b, NN * 256);
    hipMemsetAsync(q, 0, (size_t)NN * 256 * 4, stream);
    k_prop<<<NE, T, 0, stream>>>(ei, nrm, p2, q, 256);
    k_combine<<<((NN * 256) + T - 1) / T, T, 0, stream>>>(q, h2, t2, t2b, (size_t)NN * 256);

    k_gemm<<<gemm_grid(NN, 128), T, 0, stream>>>(h2b, W30t, h3, NN, 128, 256, 0);
    k_gemm<<<gemm_grid(NN, 128), T, 0, stream>>>(p2b, W31t, h3, NN, 128, 256, 1);
    k_gemm<<<gemm_grid(NN, 128), T, 0, stream>>>(t2b, W32t, h3, NN, 128, 256, 1);
    run_bn(h3, h3, (unsigned short*)nullptr, bsum, bssq, mu, rs, g3, be3, 128, 0, stream);

    // ---- FC + log_softmax
    k_fc<<<(NN + T - 1) / T, T, 0, stream>>>(h3, fcW, fcb, out, NN);
}
